// GraphMSG_87076166960229
// MI455X (gfx1250) — compile-verified, matrix operand order checked by reading, not run on hardware
//
#include <hip/hip_runtime.h>
#include <hip/hip_bf16.h>

// ---------------- problem constants ----------------
static constexpr int N_ERA = 40320;
static constexpr int N_H   = 10242;
static constexpr int E_E2H = 120960;
static constexpr int E_H2E = 120960;
static constexpr int E_H2H = 40968;
static constexpr int HID   = 128;

// input indices (top-level dict order; nested dicts flattened with sorted keys,
// JAX pytree convention: mlp -> b1,b2,beta,g,w1,w2 ; mapper -> dst,edge,msg,node,(out),src)
static constexpr int IN_X = 0, IN_XK = 1, IN_ERALL = 2, IN_HLL = 3, IN_ERATR = 4, IN_HTR = 5;
static constexpr int IN_E2H_ATTR = 6, IN_H2E_ATTR = 7, IN_H2H_ATTR = 8;
static constexpr int IN_E2H_TR = 9, IN_H2E_TR = 10, IN_H2H_TR = 11;
static constexpr int ENC_DST = 12, ENC_EDGE = 18, ENC_MSG = 24, ENC_NODE = 30, ENC_SRC = 36;
static constexpr int DEC_DST = 42, DEC_EDGE = 48, DEC_MSG = 54, DEC_NODE = 60, DEC_OUT = 66, DEC_SRC = 70;
static constexpr int PROC_EMB = 76, PROC_EDGE = 82, PROC_NODE = 88;
static constexpr int IN_KSTEP = 94;
static constexpr int IN_E2H_SRC = 95, IN_E2H_DST = 96, IN_H2E_SRC = 97, IN_H2E_DST = 98;
static constexpr int IN_H2H_SRC = 99, IN_H2H_DST = 100;

// ---------------- WMMA types ----------------
typedef __attribute__((ext_vector_type(16))) __bf16 v16bf;
typedef __attribute__((ext_vector_type(8)))  float  v8f;

union FragBF { v16bf v; uint4 q[2]; };

// =====================================================================
// Fused 2-layer MLP:  y = LN?( silu(X@W1+b1) @ W2 + b2 ) [+resid]
// Block = 64 rows x 128 cols, 8 waves, wave w owns cols [16w,16w+16).
// Weights are bf16 W^T [N][Kpad] (row n = column n of W), Kpad % 32 == 0.
// Epilogue flags: outF (plain f32 store), outB (bf16 mirror),
//                 outS+sidx (fp32 atomic scatter-add rows -> outS[sidx[row]]).
// =====================================================================
__global__ void __launch_bounds__(256) k_mlp2(
    const __hip_bfloat16* __restrict__ Xg, int M, int K1p,
    const __hip_bfloat16* __restrict__ W1Tg, const float* __restrict__ b1,
    const __hip_bfloat16* __restrict__ W2Tg, const float* __restrict__ b2, int N2,
    const float* __restrict__ g, const float* __restrict__ beta,
    const float* __restrict__ resid,
    float* __restrict__ outF, __hip_bfloat16* __restrict__ outB,
    float* __restrict__ outS, const int* __restrict__ sidx)
{
  extern __shared__ char smem[];
  __bf16* sX  = (__bf16*)smem;                                      // 64 x K1p
  __bf16* sH  = (__bf16*)(smem + (size_t)64 * K1p * 2);             // 64 x 128
  float*  sY  = (float*) (smem + (size_t)64 * K1p * 2 + 64 * 128 * 2); // 64 x 128
  float*  sMu = sY + 64 * 128;
  float*  sRs = sMu + 64;

  const __bf16* X  = reinterpret_cast<const __bf16*>(Xg);
  const __bf16* W1 = reinterpret_cast<const __bf16*>(W1Tg);
  const __bf16* W2 = reinterpret_cast<const __bf16*>(W2Tg);

  const int tid  = threadIdx.x;
  const int lane = tid & 31;
  const int wave = tid >> 5;            // 0..7
  const int lm   = lane & 15;
  const int hi   = lane >> 4;           // 0/1: which K-half this lane holds
  const int koff = hi ? 8 : 0;          // A/B frag: lane<16 -> K{0..7,16..23}, else {8..15,24..31}
  const int colBase = wave * 16;
  const int rowBase = blockIdx.x * 64;

  // ---- stage X tile (64 x K1p bf16) into LDS, zero-padded past M ----
  {
    const int rowChunks = K1p >> 3;        // 8 bf16 (16B) per chunk
    const int total = 64 * rowChunks;
    for (int c = tid; c < total; c += 256) {
      int r = c / rowChunks, cc = c - r * rowChunks;
      int grow = rowBase + r;
      uint4 v = {0u, 0u, 0u, 0u};
      if (grow < M) v = *(const uint4*)(X + (size_t)grow * K1p + cc * 8);
      *(uint4*)(sX + (size_t)r * K1p + cc * 8) = v;
    }
  }
  __syncthreads();

  v8f acc[4];
#pragma unroll
  for (int r = 0; r < 4; ++r) { v8f z = {0.f,0.f,0.f,0.f,0.f,0.f,0.f,0.f}; acc[r] = z; }

  // ---- layer 1: h = silu(X @ W1 + b1), N1 = 128 (all 8 waves active) ----
  for (int k0 = 0; k0 < K1p; k0 += 32) {
    FragBF bf;
    const __bf16* wp = W1 + (size_t)(colBase + lm) * K1p + k0 + koff;
    bf.q[0] = *(const uint4*)(wp);
    bf.q[1] = *(const uint4*)(wp + 16);
#pragma unroll
    for (int r = 0; r < 4; ++r) {
      FragBF af;
      const __bf16* ap = sX + (size_t)(r * 16 + lm) * K1p + k0 + koff;
      af.q[0] = *(const uint4*)(ap);
      af.q[1] = *(const uint4*)(ap + 16);
      acc[r] = __builtin_amdgcn_wmma_f32_16x16x32_bf16(
          false, af.v, false, bf.v, (short)0, acc[r], false, false);
    }
  }

  // epilogue 1: bias + silu -> sH (bf16). C/D layout: VGPR j -> row j (+8 for hi lanes)
#pragma unroll
  for (int r = 0; r < 4; ++r) {
#pragma unroll
    for (int j = 0; j < 8; ++j) {
      int row = r * 16 + j + hi * 8;
      int col = colBase + lm;
      float v = acc[r][j] + b1[col];
      v = v / (1.f + __expf(-v));       // silu
      sH[row * 128 + col] = (__bf16)v;
    }
  }
  __syncthreads();

  // ---- layer 2: y = h @ W2 + b2, K = 128 ----
  const bool active = colBase < N2;
#pragma unroll
  for (int r = 0; r < 4; ++r) { v8f z = {0.f,0.f,0.f,0.f,0.f,0.f,0.f,0.f}; acc[r] = z; }
  if (active) {
#pragma unroll
    for (int k0 = 0; k0 < 128; k0 += 32) {
      FragBF bf;
      const __bf16* wp = W2 + (size_t)(colBase + lm) * 128 + k0 + koff;
      bf.q[0] = *(const uint4*)(wp);
      bf.q[1] = *(const uint4*)(wp + 16);
#pragma unroll
      for (int r = 0; r < 4; ++r) {
        FragBF af;
        const __bf16* ap = sH + (size_t)(r * 16 + lm) * 128 + k0 + koff;
        af.q[0] = *(const uint4*)(ap);
        af.q[1] = *(const uint4*)(ap + 16);
        acc[r] = __builtin_amdgcn_wmma_f32_16x16x32_bf16(
            false, af.v, false, bf.v, (short)0, acc[r], false, false);
      }
    }
#pragma unroll
    for (int r = 0; r < 4; ++r)
#pragma unroll
      for (int j = 0; j < 8; ++j) {
        int row = r * 16 + j + hi * 8;
        int col = colBase + lm;
        sY[row * 128 + col] = acc[r][j] + b2[col];
      }
  }
  __syncthreads();

  // ---- optional LayerNorm over N2 cols ----
  if (g != nullptr) {
    if (tid < 64) {
      float mu = 0.f;
      for (int c = 0; c < N2; ++c) mu += sY[tid * 128 + c];
      mu /= (float)N2;
      float var = 0.f;
      for (int c = 0; c < N2; ++c) { float d = sY[tid * 128 + c] - mu; var += d * d; }
      var /= (float)N2;
      sMu[tid] = mu;
      sRs[tid] = rsqrtf(var + 1e-5f);
    }
    __syncthreads();
  }

  // ---- output epilogue ----
  const int total = 64 * N2;
  for (int i = tid; i < total; i += 256) {
    int row = i / N2, col = i - row * N2;
    int grow = rowBase + row;
    if (grow >= M) continue;
    float v = sY[row * 128 + col];
    if (g)     v = (v - sMu[row]) * sRs[row] * g[col] + beta[col];
    if (resid) v += resid[(size_t)grow * N2 + col];
    if (outF)  outF[(size_t)grow * N2 + col] = v;
    if (outB)  outB[(size_t)grow * N2 + col] = __float2bfloat16(v);
    if (outS)  atomicAdd(outS + (size_t)sidx[grow] * N2 + col, v);
  }
}

// ---- weight fp32[K][N] -> bf16 W^T[N][Kpad] ----
__global__ void k_wt(const float* __restrict__ w, __hip_bfloat16* __restrict__ wt,
                     int K, int N, int Kp) {
  int i = blockIdx.x * 256 + threadIdx.x;
  if (i >= N * Kp) return;
  int n = i / Kp, k = i - n * Kp;
  wt[i] = __float2bfloat16(k < K ? w[(size_t)k * N + n] : 0.f);
}

// ---- x_era feature build: [x(m,f)=144 | kstep | xk:64 | latlon:4 | train:8 | pad] -> 224 bf16 ----
__global__ void k_build_xera(const float* __restrict__ x, const float* __restrict__ xk,
                             const float* __restrict__ ll, const float* __restrict__ tr,
                             const int* __restrict__ kstep, __hip_bfloat16* __restrict__ out) {
  int i = blockIdx.x * 256 + threadIdx.x;
  if (i >= N_ERA * 224) return;
  int n = i / 224, c = i - n * 224;
  float v = 0.f;
  if      (c < 144) v = x[((size_t)(c / 72) * N_ERA + n) * 72 + (c % 72)];
  else if (c == 144) v = (float)kstep[0];
  else if (c < 209) v = xk[(size_t)n * 64 + (c - 145)];
  else if (c < 213) v = ll[(size_t)n * 4 + (c - 209)];
  else if (c < 221) v = tr[(size_t)n * 8 + (c - 213)];
  out[i] = __float2bfloat16(v);
}

// ---- [a:ca | b:cb | pad] -> 32 bf16 per row ----
__global__ void k_build_feat(const float* __restrict__ a, int ca,
                             const float* __restrict__ b, int cb,
                             int M, __hip_bfloat16* __restrict__ out) {
  int i = blockIdx.x * 256 + threadIdx.x;
  if (i >= M * 32) return;
  int m = i / 32, c = i - m * 32;
  float v = 0.f;
  if      (c < ca)      v = a[(size_t)m * ca + c];
  else if (c < ca + cb) v = b[(size_t)m * cb + (c - ca)];
  out[i] = __float2bfloat16(v);
}

// ---- msg input: out[E,384] = [s[si[e]] | d[di[e]] | e[e]] (bf16, 16B chunks) ----
__global__ void k_concat3(const __hip_bfloat16* __restrict__ s, const int* __restrict__ si,
                          const __hip_bfloat16* __restrict__ d, const int* __restrict__ di,
                          const __hip_bfloat16* __restrict__ e, int E,
                          __hip_bfloat16* __restrict__ out) {
  int i = blockIdx.x * 256 + threadIdx.x;
  if (i >= E * 48) return;
  int ed = i / 48, c = i - ed * 48;
  const __hip_bfloat16* src;
  if      (c < 16) src = s + (size_t)si[ed] * 128 + c * 8;
  else if (c < 32) src = d + (size_t)di[ed] * 128 + (c - 16) * 8;
  else             src = e + (size_t)ed * 128 + (c - 32) * 8;
  *(uint4*)(out + (size_t)ed * 384 + c * 8) = *(const uint4*)src;
}

// ---- node-update input: out[M,256] = [a_bf16 | bf16(b_f32)] ----
__global__ void k_concat2(const __hip_bfloat16* __restrict__ a, const float* __restrict__ bF,
                          int M, __hip_bfloat16* __restrict__ out) {
  int i = blockIdx.x * 256 + threadIdx.x;
  if (i >= M * 256) return;
  int m = i / 256, c = i - m * 256;
  out[i] = (c < 128) ? a[(size_t)m * 128 + c]
                     : __float2bfloat16(bF[(size_t)m * 128 + (c - 128)]);
}

__global__ void k_zero(float* __restrict__ p, int n) {
  int i = blockIdx.x * 256 + threadIdx.x;
  if (i < n) p[i] = 0.f;
}

// =====================================================================
struct MlpW {
  __hip_bfloat16 *w1, *w2;
  const float *b1, *b2, *g, *beta;
  int K1p, N2;
};

extern "C" void kernel_launch(void* const* d_in, const int* in_sizes, int n_in,
                              void* d_out, int out_size, void* d_ws, size_t ws_size,
                              hipStream_t stream) {
  (void)in_sizes; (void)n_in; (void)out_size; (void)ws_size;
  auto F = [&](int i) { return (const float*)d_in[i]; };
  auto I = [&](int i) { return (const int*)d_in[i]; };

  size_t off = 0;
  auto alloc  = [&](size_t bytes) -> void* {
    void* p = (char*)d_ws + off;
    off += (bytes + 255) & ~(size_t)255;
    return p;
  };
  auto allocB = [&](size_t n) { return (__hip_bfloat16*)alloc(n * 2); };
  auto allocF = [&](size_t n) { return (float*)alloc(n * 4); };

  auto cvtW = [&](const float* w, int K, int N, int Kp) {
    __hip_bfloat16* wt = allocB((size_t)N * Kp);
    int tot = N * Kp;
    k_wt<<<(tot + 255) / 256, 256, 0, stream>>>(w, wt, K, N, Kp);
    return wt;
  };

  auto mkMlp = [&](int i0, int K1, int Kp, int N2out, bool ln) -> MlpW {
    MlpW m{};
    m.b1 = F(i0); m.b2 = F(i0 + 1);
    const float *w1, *w2;
    if (ln) { m.beta = F(i0 + 2); m.g = F(i0 + 3); w1 = F(i0 + 4); w2 = F(i0 + 5); }
    else    { m.beta = nullptr;   m.g = nullptr;   w1 = F(i0 + 2); w2 = F(i0 + 3); }
    m.w1 = cvtW(w1, K1, HID, Kp);
    m.w2 = cvtW(w2, HID, N2out, HID);
    m.K1p = Kp; m.N2 = N2out;
    return m;
  };
  auto mkMlpStack = [&](int i0, int K1, int Kp, int l) -> MlpW {   // proc layers (stacked)
    MlpW m{};
    m.b1 = F(i0) + l * HID; m.b2 = F(i0 + 1) + l * HID;
    m.beta = F(i0 + 2) + l * HID; m.g = F(i0 + 3) + l * HID;
    m.w1 = cvtW(F(i0 + 4) + (size_t)l * K1 * HID, K1, HID, Kp);
    m.w2 = cvtW(F(i0 + 5) + (size_t)l * HID * HID, HID, HID, HID);
    m.K1p = Kp; m.N2 = HID;
    return m;
  };

  auto runMlp = [&](const MlpW& m, const __hip_bfloat16* X, int M,
                    const float* resid, float* outF, __hip_bfloat16* outB,
                    float* outS, const int* sidx) {
    int blocks = (M + 63) / 64;
    size_t shmem = (size_t)64 * m.K1p * 2 + 64 * 128 * 2 + 64 * 128 * 4 + 64 * 2 * 4;
    k_mlp2<<<blocks, 256, shmem, stream>>>(X, M, m.K1p, m.w1, m.b1, m.w2, m.b2,
                                           m.N2, m.g, m.beta, resid, outF, outB, outS, sidx);
  };

  // ---- weights (bf16, transposed, K padded to 32) ----
  MlpW encdst  = mkMlp(ENC_DST, 12, 32, HID, true);
  MlpW encedge = mkMlp(ENC_EDGE, 13, 32, HID, true);
  MlpW encmsg  = mkMlp(ENC_MSG, 384, 384, HID, true);
  MlpW encnode = mkMlp(ENC_NODE, 256, 256, HID, true);
  MlpW encsrc  = mkMlp(ENC_SRC, 221, 224, HID, true);
  MlpW decdst  = mkMlp(DEC_DST, 128, 128, HID, true);
  MlpW decedge = mkMlp(DEC_EDGE, 13, 32, HID, true);
  MlpW decmsg  = mkMlp(DEC_MSG, 384, 384, HID, true);
  MlpW decnode = mkMlp(DEC_NODE, 256, 256, HID, true);
  MlpW decout  = mkMlp(DEC_OUT, 128, 128, 64, false);
  MlpW decsrc  = mkMlp(DEC_SRC, 128, 128, HID, true);
  MlpW procemb = mkMlp(PROC_EMB, 13, 32, HID, true);
  MlpW procedge[2] = { mkMlpStack(PROC_EDGE, 384, 384, 0), mkMlpStack(PROC_EDGE, 384, 384, 1) };
  MlpW procnode[2] = { mkMlpStack(PROC_NODE, 256, 256, 0), mkMlpStack(PROC_NODE, 256, 256, 1) };

  // ---- activation buffers ----
  __hip_bfloat16* xera   = allocB((size_t)N_ERA * 224);
  __hip_bfloat16* xhin   = allocB((size_t)N_H * 32);
  __hip_bfloat16* edgein = allocB((size_t)E_E2H * 32);   // reused: e2h then h2e
  __hip_bfloat16* h2hin  = allocB((size_t)E_H2H * 32);
  __hip_bfloat16* srcemb = allocB((size_t)N_ERA * 128);  // x_era_emb
  float*          xh_f   = allocF((size_t)N_H * 128);
  __hip_bfloat16* xh_b   = allocB((size_t)N_H * 128);
  __hip_bfloat16* eemb   = allocB((size_t)E_E2H * 128);  // reused: enc/dec edge emb
  __hip_bfloat16* msgbuf = allocB((size_t)E_E2H * 384);  // reused by all msg MLPs
  float*          agg    = allocF((size_t)N_ERA * 128);  // reused; zeroed per use
  __hip_bfloat16* cat2   = allocB((size_t)N_ERA * 256);  // reused node-update input
  float*          e_f    = allocF((size_t)E_H2H * 128);
  __hip_bfloat16* e_b    = allocB((size_t)E_H2H * 128);
  __hip_bfloat16* dsrc_b = allocB((size_t)N_H * 128);
  float*          xera2f = allocF((size_t)N_ERA * 128);
  __hip_bfloat16* xera2b = allocB((size_t)N_ERA * 128);

  // ---- features ----
  k_build_xera<<<(N_ERA * 224 + 255) / 256, 256, 0, stream>>>(
      F(IN_X), F(IN_XK), F(IN_ERALL), F(IN_ERATR), I(IN_KSTEP), xera);
  k_build_feat<<<(N_H * 32 + 255) / 256, 256, 0, stream>>>(F(IN_HLL), 4, F(IN_HTR), 8, N_H, xhin);
  k_build_feat<<<(E_E2H * 32 + 255) / 256, 256, 0, stream>>>(F(IN_E2H_ATTR), 3, F(IN_E2H_TR), 8, E_E2H, edgein);
  k_build_feat<<<(E_H2H * 32 + 255) / 256, 256, 0, stream>>>(F(IN_H2H_ATTR), 3, F(IN_H2H_TR), 8, E_H2H, h2hin);

  // ---- encoder (era -> h) ----
  runMlp(encsrc, xera, N_ERA, nullptr, nullptr, srcemb, nullptr, nullptr);
  runMlp(encdst, xhin, N_H, nullptr, xh_f, xh_b, nullptr, nullptr);
  runMlp(encedge, edgein, E_E2H, nullptr, nullptr, eemb, nullptr, nullptr);
  k_concat3<<<(E_E2H * 48 + 255) / 256, 256, 0, stream>>>(
      srcemb, I(IN_E2H_SRC), xh_b, I(IN_E2H_DST), eemb, E_E2H, msgbuf);
  k_zero<<<(N_H * 128 + 255) / 256, 256, 0, stream>>>(agg, N_H * 128);
  runMlp(encmsg, msgbuf, E_E2H, nullptr, nullptr, nullptr, agg, I(IN_E2H_DST));
  k_concat2<<<(N_H * 256 + 255) / 256, 256, 0, stream>>>(xh_b, agg, N_H, cat2);
  runMlp(encnode, cat2, N_H, xh_f, xh_f, xh_b, nullptr, nullptr);

  // ---- processor (h -> h, 2 layers, residual edge+node) ----
  runMlp(procemb, h2hin, E_H2H, nullptr, e_f, e_b, nullptr, nullptr);
  for (int l = 0; l < 2; ++l) {
    k_concat3<<<(E_H2H * 48 + 255) / 256, 256, 0, stream>>>(
        xh_b, I(IN_H2H_SRC), xh_b, I(IN_H2H_DST), e_b, E_H2H, msgbuf);
    k_zero<<<(N_H * 128 + 255) / 256, 256, 0, stream>>>(agg, N_H * 128);
    // e = e + MLP(...); simultaneously scatter-add e_new into agg (segment_sum)
    runMlp(procedge[l], msgbuf, E_H2H, e_f, e_f, e_b, agg, I(IN_H2H_DST));
    k_concat2<<<(N_H * 256 + 255) / 256, 256, 0, stream>>>(xh_b, agg, N_H, cat2);
    runMlp(procnode[l], cat2, N_H, xh_f, xh_f, xh_b, nullptr, nullptr);
  }

  // ---- decoder (h -> era) ----
  runMlp(decsrc, xh_b, N_H, nullptr, nullptr, dsrc_b, nullptr, nullptr);
  runMlp(decdst, srcemb, N_ERA, nullptr, xera2f, xera2b, nullptr, nullptr);
  k_build_feat<<<(E_H2E * 32 + 255) / 256, 256, 0, stream>>>(F(IN_H2E_ATTR), 3, F(IN_H2E_TR), 8, E_H2E, edgein);
  runMlp(decedge, edgein, E_H2E, nullptr, nullptr, eemb, nullptr, nullptr);
  k_concat3<<<(E_H2E * 48 + 255) / 256, 256, 0, stream>>>(
      dsrc_b, I(IN_H2E_SRC), xera2b, I(IN_H2E_DST), eemb, E_H2E, msgbuf);
  k_zero<<<(N_ERA * 128 + 255) / 256, 256, 0, stream>>>(agg, N_ERA * 128);
  runMlp(decmsg, msgbuf, E_H2E, nullptr, nullptr, nullptr, agg, I(IN_H2E_DST));
  k_concat2<<<(N_ERA * 256 + 255) / 256, 256, 0, stream>>>(xera2b, agg, N_ERA, cat2);
  runMlp(decnode, cat2, N_ERA, xera2f, xera2f, xera2b, nullptr, nullptr);

  // ---- output projection: (40320, 128) -> (40320, 64) fp32 ----
  runMlp(decout, xera2b, N_ERA, nullptr, (float*)d_out, nullptr, nullptr, nullptr);
}
